// LRNetLinear_15058155339868
// MI455X (gfx1250) — compile-verified, matrix operand order checked by reading, not run on hardware
//
#include <hip/hip_runtime.h>
#include <hip/hip_bf16.h>
#include <stdint.h>

#define O_DIM 4096
#define I_DIM 4096
#define B_DIM 4096

#define BM 128   // tokens per block tile
#define BN 64    // out-features per block tile
#define BK 32    // K step (matches wmma 16x16x32)
#define PK 40    // padded LDS row stride in f16 (80B, 16B-aligned, conflict-avoiding)

typedef __attribute__((ext_vector_type(16))) _Float16 v16h;
typedef __attribute__((ext_vector_type(8)))  _Float16 v8h;
typedef __attribute__((ext_vector_type(4)))  _Float16 v4h;
typedef __attribute__((ext_vector_type(8)))  float    v8f;
typedef __attribute__((ext_vector_type(4)))  float    v4f;
typedef __attribute__((ext_vector_type(4)))  int      v4i;

// ---- async global->LDS path (CDNA5), with portable fallback -----------------
#if defined(__has_builtin)
#  if __has_builtin(__builtin_amdgcn_global_load_async_to_lds_b128) && \
      __has_builtin(__builtin_amdgcn_s_wait_asynccnt)
#    define USE_ASYNC_LDS 1
#  endif
#endif
#ifndef USE_ASYNC_LDS
#  define USE_ASYNC_LDS 0
#endif

#if USE_ASYNC_LDS
typedef __attribute__((address_space(1))) v4i* gv4i_ptr;   // global int4*
typedef __attribute__((address_space(3))) v4i* sv4i_ptr;   // LDS int4*
#  define WAIT_ASYNC(n) __builtin_amdgcn_s_wait_asynccnt(n)
#else
#  define WAIT_ASYNC(n)
#endif

__device__ __forceinline__ void cp16(const _Float16* g, _Float16* l) {
#if USE_ASYNC_LDS
  __builtin_amdgcn_global_load_async_to_lds_b128(
      (gv4i_ptr)(void*)g, (sv4i_ptr)(void*)l, 0, 0);
#else
  *(v8h*)l = *(const v8h*)g;
#endif
}

// ---- prep: x -> f16, x*x -> f16 --------------------------------------------
__global__ void lrnet_prep_x(const float* __restrict__ x,
                             _Float16* __restrict__ xh,
                             _Float16* __restrict__ x2h) {
  long idx = ((long)blockIdx.x * blockDim.x + threadIdx.x) * 4;
  v4f v = *(const v4f*)(x + idx);
  v4h a, b;
#pragma unroll
  for (int i = 0; i < 4; ++i) {
    a[i] = (_Float16)v[i];
    b[i] = (_Float16)(v[i] * v[i]);
  }
  *(v4h*)(xh + idx)  = a;
  *(v4h*)(x2h + idx) = b;
}

// ---- prep: 3-way softmax -> w_mean (f16), w_var (f16) ----------------------
__global__ void lrnet_prep_w(const float* __restrict__ tn,
                             const float* __restrict__ tp,
                             const float* __restrict__ sc,
                             _Float16* __restrict__ wm,
                             _Float16* __restrict__ wv) {
  long idx = ((long)blockIdx.x * blockDim.x + threadIdx.x) * 4;
  v4f n = *(const v4f*)(tn + idx);
  v4f p = *(const v4f*)(tp + idx);
  v4f s = *(const v4f*)(sc + idx);
  v4h m4, v4;
#pragma unroll
  for (int i = 0; i < 4; ++i) {
    float mx  = fmaxf(fmaxf(n[i], p[i]), 0.0f);
    float en  = __expf(n[i] - mx);
    float ez  = __expf(-mx);
    float ep  = __expf(p[i] - mx);
    float inv = 1.0f / (en + ez + ep);
    float pn  = en * inv;
    float pp  = ep * inv;
    float d   = pp - pn;
    m4[i] = (_Float16)(d * s[i]);
    v4[i] = (_Float16)((pp + pn - d * d) * (s[i] * s[i]));
  }
  *(v4h*)(wm + idx) = m4;
  *(v4h*)(wv + idx) = v4;
}

// ---- fused dual-GEMM + epilogue --------------------------------------------
// mu = Xh @ Wm^T ; var = X2h @ Wv^T ; out = mu + sqrt(max(var,1e-8))*eps
__global__ __launch_bounds__(256)
void lrnet_gemm_fused(const _Float16* __restrict__ xh,
                      const _Float16* __restrict__ x2h,
                      const _Float16* __restrict__ wm,
                      const _Float16* __restrict__ wv,
                      const float* __restrict__ eps,
                      float* __restrict__ out) {
  // LDS layout (f16 units), double-buffered:
  //   sAx  [2][BM*PK] = 10240   sAx2 [2][BM*PK] = 10240
  //   sBm  [2][BN*PK] =  5120   sBv  [2][BN*PK] =  5120      total 61440 B
  __shared__ _Float16 lds[30720];
  _Float16* sAx  = lds;
  _Float16* sAx2 = lds + 10240;
  _Float16* sBm  = lds + 20480;
  _Float16* sBv  = lds + 25600;

  const int  tid  = threadIdx.x;
  const int  lane = tid & 31;
  const int  wid  = tid >> 5;          // 8 waves
  const int  wmi  = wid >> 1;          // 0..3 (M slot, 32 rows each)
  const int  wni  = wid & 1;           // 0..1 (N slot, 32 cols each)
  const int  lr   = lane & 15;
  const int  lh   = lane >> 4;

  const long rowA = (long)blockIdx.y * BM;   // token base
  const long rowB = (long)blockIdx.x * BN;   // out-feature base

  auto load_stage = [&](int st, int kt) {
    const long k0 = (long)kt * BK;
    // A tiles (x and x^2): 128x32 f16 = 512 16B-chunks each
#pragma unroll
    for (int j = 0; j < 2; ++j) {
      int  c  = tid + j * 256;
      int  r  = c >> 2;
      int  kc = (c & 3) * 8;
      long g  = (rowA + r) * (long)I_DIM + k0 + kc;
      int  lo = st * 5120 + r * PK + kc;
      cp16(xh  + g, sAx  + lo);
      cp16(x2h + g, sAx2 + lo);
    }
    // B tiles (w_mean and w_var): 64x32 f16 = 256 16B-chunks each
    {
      int  r  = tid >> 2;
      int  kc = (tid & 3) * 8;
      long g  = (rowB + r) * (long)I_DIM + k0 + kc;
      int  lo = st * 2560 + r * PK + kc;
      cp16(wm + g, sBm + lo);
      cp16(wv + g, sBv + lo);
    }
  };

  v8f accM[2][2] = {};
  v8f accV[2][2] = {};

  const int KT = I_DIM / BK;   // 128
  load_stage(0, 0);

  for (int kt = 0; kt < KT; ++kt) {
    const int st = kt & 1;
    if (kt + 1 < KT) {
      load_stage(st ^ 1, kt + 1);
      // L2 prefetch a few stages ahead
      if (kt + 4 < KT) {
        __builtin_prefetch(xh + (rowA + (tid >> 2)) * (long)I_DIM + (long)(kt + 4) * BK, 0, 0);
      }
      WAIT_ASYNC(6);   // 6 async ops outstanding => stage kt landed
    } else {
      WAIT_ASYNC(0);
    }
    __syncthreads();

    const _Float16* pAx  = sAx  + st * 5120;
    const _Float16* pAx2 = sAx2 + st * 5120;
    const _Float16* pBm  = sBm  + st * 2560;
    const _Float16* pBv  = sBv  + st * 2560;

    union F16x16 { v16h v; v8h h[2]; };
    F16x16 ax[2], ax2[2], fbm[2], fbv[2];

    // A fragment (16-bit 16x32): lane = (M & 15) + 16*(Khalf)
#pragma unroll
    for (int mi = 0; mi < 2; ++mi) {
      const int r = wmi * 32 + mi * 16 + lr;
      ax[mi].h[0]  = *(const v8h*)(pAx  + r * PK + lh * 8);
      ax[mi].h[1]  = *(const v8h*)(pAx  + r * PK + 16 + lh * 8);
      ax2[mi].h[0] = *(const v8h*)(pAx2 + r * PK + lh * 8);
      ax2[mi].h[1] = *(const v8h*)(pAx2 + r * PK + 16 + lh * 8);
    }
    // B fragment (32x16): per-lane K = 8*lh+s, 16+8*lh+(s-8) (same order as A) from row-major [O, I] weights
#pragma unroll
    for (int ni = 0; ni < 2; ++ni) {
      const int r = wni * 32 + ni * 16 + lr;
      fbm[ni].h[0] = *(const v8h*)(pBm + r * PK + lh * 8);
      fbm[ni].h[1] = *(const v8h*)(pBm + r * PK + 16 + lh * 8);
      fbv[ni].h[0] = *(const v8h*)(pBv + r * PK + lh * 8);
      fbv[ni].h[1] = *(const v8h*)(pBv + r * PK + 16 + lh * 8);
    }

#pragma unroll
    for (int mi = 0; mi < 2; ++mi)
#pragma unroll
      for (int ni = 0; ni < 2; ++ni) {
        accM[mi][ni] = __builtin_amdgcn_wmma_f32_16x16x32_f16(
            false, ax[mi].v,  false, fbm[ni].v, (short)0, accM[mi][ni], false, false);
        accV[mi][ni] = __builtin_amdgcn_wmma_f32_16x16x32_f16(
            false, ax2[mi].v, false, fbv[ni].v, (short)0, accV[mi][ni], false, false);
      }

    __syncthreads();   // protect buffer st before it is refilled at kt+1
  }

  // Epilogue: C layout is n = lane&15, m = 8*(lane>>4) + e
#pragma unroll
  for (int mi = 0; mi < 2; ++mi)
#pragma unroll
    for (int ni = 0; ni < 2; ++ni) {
      const long o = rowB + wni * 32 + ni * 16 + lr;
#pragma unroll
      for (int e = 0; e < 8; ++e) {
        const long b   = rowA + wmi * 32 + mi * 16 + lh * 8 + e;
        const long idx = b * (long)O_DIM + o;
        float mu  = accM[mi][ni][e];
        float var = accV[mi][ni][e];
        float sig = sqrtf(fmaxf(var, 1e-8f));
        out[idx] = mu + sig * eps[idx];
      }
    }
}

// ---- launcher ---------------------------------------------------------------
extern "C" void kernel_launch(void* const* d_in, const int* in_sizes, int n_in,
                              void* d_out, int out_size, void* d_ws, size_t ws_size,
                              hipStream_t stream) {
  const float* x   = (const float*)d_in[0];
  const float* tn  = (const float*)d_in[1];
  const float* tp  = (const float*)d_in[2];
  const float* sc  = (const float*)d_in[3];
  const float* eps = (const float*)d_in[4];
  float* out = (float*)d_out;

  _Float16* xh  = (_Float16*)d_ws;                       // 32 MB
  _Float16* x2h = xh  + (size_t)B_DIM * I_DIM;           // 32 MB
  _Float16* wmn = x2h + (size_t)B_DIM * I_DIM;           // 32 MB
  _Float16* wvr = wmn + (size_t)O_DIM * I_DIM;           // 32 MB

  {
    const long n4 = (long)B_DIM * I_DIM / 4;             // 4 elems/thread
    lrnet_prep_x<<<dim3((unsigned)(n4 / 256)), 256, 0, stream>>>(x, xh, x2h);
  }
  {
    const long n4 = (long)O_DIM * I_DIM / 4;
    lrnet_prep_w<<<dim3((unsigned)(n4 / 256)), 256, 0, stream>>>(tn, tp, sc, wmn, wvr);
  }
  dim3 grid(O_DIM / BN, B_DIM / BM);   // 64 x 32 blocks
  lrnet_gemm_fused<<<grid, 256, 0, stream>>>(xh, x2h, wmn, wvr, eps, out);
}
